// SoftMoELayerWrapper_5042291606062
// MI455X (gfx1250) — compile-verified, matrix operand order checked by reading, not run on hardware
//
#include <hip/hip_runtime.h>

// ---------------------------------------------------------------------------
// Soft-MoE forward for MI455X (gfx1250, wave32, WMMA bf16 -> f32 accumulate)
// B=64, C=D=384, H=W=32 (M=1024 tokens), N=128 experts, 1 slot/expert.
// GEMM waves use 2x2 register tiles (32x32 output) for 2x operand reuse.
// ---------------------------------------------------------------------------

typedef __attribute__((ext_vector_type(8)))  __bf16 bf16x8;
typedef __attribute__((ext_vector_type(16))) __bf16 bf16x16;
typedef __attribute__((ext_vector_type(8)))  float  floatx8;

#define BATCH 64
#define DIM   384
#define MTOK  1024
#define NEXP  128
#define EPS_N 1e-12f

#define WMMA_BF16(A, B, C) \
    __builtin_amdgcn_wmma_f32_16x16x32_bf16(false, (A), false, (B), (short)0, (C), false, false)

// Load 16 bf16 for one lane's slice of a 16x32 WMMA operand.
// `base` already includes the (lane>=16 ? 8 : 0) K offset; the second 8-wide
// chunk sits 16 K-elements further along the row.
__device__ __forceinline__ bf16x16 load_k32(const __bf16* __restrict__ base) {
    bf16x8 lo = *reinterpret_cast<const bf16x8*>(base);
    bf16x8 hi = *reinterpret_cast<const bf16x8*>(base + 16);
    return __builtin_shufflevector(lo, hi, 0,1,2,3,4,5,6,7,8,9,10,11,12,13,14,15);
}

__device__ __forceinline__ void store_tile_f32(float* __restrict__ base, int ldc,
                                               const floatx8& acc) {
    #pragma unroll
    for (int r = 0; r < 8; ++r) base[(size_t)r * ldc] = acc[r];
}

__device__ __forceinline__ void store_tile_bf16(__bf16* __restrict__ base, int ldc,
                                                const floatx8& acc) {
    #pragma unroll
    for (int r = 0; r < 8; ++r) base[(size_t)r * ldc] = (__bf16)acc[r];
}

// ---------------------------------------------------------------------------
// Kernel 1: token L2-normalize + transpose. x[B][C][M] (f32) ->
//   xn_md[B][M][D] bf16  (token-major, A-operand of logits GEMM)
//   xn_dm[B][D][M] bf16  (channel-major, B-operand of slot GEMM)
// One block = 32 tokens of one batch. LDS tile [32][384] f32, padded.
// ---------------------------------------------------------------------------
__global__ void k_norm_tokens(const float* __restrict__ x,
                              __bf16* __restrict__ xn_md,
                              __bf16* __restrict__ xn_dm) {
    __shared__ float tile[32 * 385];
    __shared__ float part[32][8];
    __shared__ float invn[32];
    const int b  = blockIdx.y;
    const int m0 = blockIdx.x * 32;
    const int t  = threadIdx.x;          // 256 threads
    const int tok  = t & 31;
    const int crow = t >> 5;             // 0..7
    const float* xb = x + (size_t)b * DIM * MTOK;
    for (int cb = 0; cb < DIM; cb += 8) {
        int c = cb + crow;
        tile[tok * 385 + c] = xb[(size_t)c * MTOK + m0 + tok];
    }
    __syncthreads();
    {   // partial sum of squares: 8 threads per token
        int tk = t >> 3, p = t & 7;
        float s = 0.f;
        for (int c = p * 48; c < p * 48 + 48; ++c) {
            float v = tile[tk * 385 + c];
            s += v * v;
        }
        part[tk][p] = s;
    }
    __syncthreads();
    if (t < 32) {
        float s = 0.f;
        #pragma unroll
        for (int p = 0; p < 8; ++p) s += part[t][p];
        invn[t] = 1.0f / fmaxf(sqrtf(s), EPS_N);
    }
    __syncthreads();
    for (int e = t; e < 32 * DIM; e += 256) {           // [M][D], coalesced in d
        int tk = e / DIM, d = e - tk * DIM;
        float v = tile[tk * 385 + d] * invn[tk];
        xn_md[((size_t)(b * MTOK + m0 + tk)) * DIM + d] = (__bf16)v;
    }
    for (int e = t; e < 32 * DIM; e += 256) {           // [D][M], coalesced in m
        int d = e >> 5, tk = e & 31;
        float v = tile[tk * 385 + d] * invn[tk];
        xn_dm[((size_t)(b * DIM + d)) * MTOK + m0 + tk] = (__bf16)v;
    }
}

// ---------------------------------------------------------------------------
// Kernel 2: phi normalize: phi[C][N][1] f32 -> phin[N][D] bf16 (row = expert)
// ---------------------------------------------------------------------------
__global__ void k_norm_phi(const float* __restrict__ phi,
                           const float* __restrict__ scale,
                           __bf16* __restrict__ phin) {
    const int n = threadIdx.x;           // 128 threads, 1 block
    float ss = 0.f;
    for (int c = 0; c < DIM; ++c) {
        float v = phi[(size_t)c * NEXP + n];
        ss += v * v;
    }
    float inv = scale[0] / fmaxf(sqrtf(ss), EPS_N);
    for (int c = 0; c < DIM; ++c)
        phin[(size_t)n * DIM + c] = (__bf16)(phi[(size_t)c * NEXP + n] * inv);
}

// ---------------------------------------------------------------------------
// Kernel 3: logits[b] = xn_md[b] (1024x384) x phin^T (384x128) -> f32
// grid (N/32, M/256, B); wave computes a 32x32 tile via 2x2 accumulators.
// ---------------------------------------------------------------------------
__global__ void k_gemm_logits(const __bf16* __restrict__ xn_md,
                              const __bf16* __restrict__ phin,
                              float* __restrict__ logits) {
    const int b    = blockIdx.z;
    const int w    = threadIdx.x >> 5;
    const int lane = threadIdx.x & 31;
    const int row  = lane & 15;
    const int half = lane >> 4;
    const int m0 = blockIdx.y * 256 + w * 32;
    const int n0 = blockIdx.x * 32;
    const __bf16* a0p = xn_md + ((size_t)(b * MTOK + m0 + row)) * DIM + half * 8;
    const __bf16* a1p = a0p + (size_t)16 * DIM;
    const __bf16* b0p = phin + ((size_t)(n0 + row)) * DIM + half * 8;
    const __bf16* b1p = b0p + (size_t)16 * DIM;
    floatx8 acc00 = {}, acc01 = {}, acc10 = {}, acc11 = {};
    #pragma unroll 2
    for (int kk = 0; kk < DIM; kk += 32) {
        bf16x16 a0 = load_k32(a0p + kk), a1 = load_k32(a1p + kk);
        bf16x16 b0 = load_k32(b0p + kk), b1 = load_k32(b1p + kk);
        acc00 = WMMA_BF16(a0, b0, acc00);
        acc01 = WMMA_BF16(a0, b1, acc01);
        acc10 = WMMA_BF16(a1, b0, acc10);
        acc11 = WMMA_BF16(a1, b1, acc11);
    }
    float* d00 = logits + ((size_t)(b * MTOK + m0 + half * 8)) * NEXP + n0 + row;
    store_tile_f32(d00,                    NEXP, acc00);
    store_tile_f32(d00 + 16,               NEXP, acc01);
    store_tile_f32(d00 + (size_t)16 * NEXP, NEXP, acc10);
    store_tile_f32(d00 + (size_t)16 * NEXP + 16, NEXP, acc11);
}

// ---------------------------------------------------------------------------
// Kernels 4a/4b/4c: split dispatch softmax over tokens (axis M).
// 4a: per-(b, 64-row chunk) column max + sum-exp partials. grid (16, B).
// 4b: combine 16 chunk partials -> global max & 1/sum. grid (B).
// 4c: normalized write, LDS-transposed so d_nm[B][N][M] is stored coalesced.
// ---------------------------------------------------------------------------
__global__ void k_smax_d_part(const float* __restrict__ logits,
                              float* __restrict__ cmax, float* __restrict__ csum) {
    const int b = blockIdx.y, ch = blockIdx.x;
    const int n = threadIdx.x;           // 128 threads
    const float* base = logits + ((size_t)b * MTOK + ch * 64) * NEXP + n;
    float mx = -3.4e38f;
    for (int m = 0; m < 64; ++m) mx = fmaxf(mx, base[(size_t)m * NEXP]);
    float s = 0.f;
    for (int m = 0; m < 64; ++m) s += __expf(base[(size_t)m * NEXP] - mx);
    cmax[((size_t)b * 16 + ch) * NEXP + n] = mx;
    csum[((size_t)b * 16 + ch) * NEXP + n] = s;
}

__global__ void k_smax_d_comb(const float* __restrict__ cmax,
                              const float* __restrict__ csum,
                              float* __restrict__ gmax, float* __restrict__ ginv) {
    const int b = blockIdx.x;
    const int n = threadIdx.x;           // 128 threads
    float gm = -3.4e38f;
    #pragma unroll
    for (int c = 0; c < 16; ++c)
        gm = fmaxf(gm, cmax[((size_t)b * 16 + c) * NEXP + n]);
    float s = 0.f;
    #pragma unroll
    for (int c = 0; c < 16; ++c)
        s += csum[((size_t)b * 16 + c) * NEXP + n] *
             __expf(cmax[((size_t)b * 16 + c) * NEXP + n] - gm);
    gmax[(size_t)b * NEXP + n] = gm;
    ginv[(size_t)b * NEXP + n] = 1.0f / s;
}

__global__ void k_smax_d_write(const float* __restrict__ logits,
                               const float* __restrict__ gmax,
                               const float* __restrict__ ginv,
                               __bf16* __restrict__ d_nm) {
    __shared__ __bf16 lds[NEXP * 65];
    const int b  = blockIdx.y;
    const int m0 = blockIdx.x * 64;
    const int t  = threadIdx.x;          // 256 threads
    const int n  = t & 127;
    const int mo = t >> 7;               // 0..1
    const float gm = gmax[(size_t)b * NEXP + n];
    const float iv = ginv[(size_t)b * NEXP + n];
    const float* base = logits + ((size_t)b * MTOK + m0) * NEXP + n;
    for (int i = 0; i < 32; ++i) {
        int m = mo + i * 2;
        float v = base[(size_t)m * NEXP];
        lds[n * 65 + m] = (__bf16)(__expf(v - gm) * iv);
    }
    __syncthreads();
    for (int e = t; e < NEXP * 64; e += 256) {
        int nn = e >> 6, mm = e & 63;
        d_nm[((size_t)(b * NEXP + nn)) * MTOK + m0 + mm] = lds[nn * 65 + mm];
    }
}

// ---------------------------------------------------------------------------
// Kernel 5: combine softmax over experts (axis N) per token. One wave/token.
// Output c_mn[B][M][N] bf16 (token rows contiguous in N for the final GEMM).
// ---------------------------------------------------------------------------
__global__ void k_softmax_combine(const float* __restrict__ logits,
                                  __bf16* __restrict__ c_mn) {
    const int w    = threadIdx.x >> 5;
    const int lane = threadIdx.x & 31;
    const int tk   = blockIdx.x * 8 + w;        // global (b*M + m)
    const float* base = logits + (size_t)tk * NEXP;
    float v[4];
    float mx = -3.4e38f;
    #pragma unroll
    for (int j = 0; j < 4; ++j) { v[j] = base[lane + j * 32]; mx = fmaxf(mx, v[j]); }
    #pragma unroll
    for (int off = 16; off > 0; off >>= 1) mx = fmaxf(mx, __shfl_xor(mx, off, 32));
    float s = 0.f;
    #pragma unroll
    for (int j = 0; j < 4; ++j) { v[j] = __expf(v[j] - mx); s += v[j]; }
    #pragma unroll
    for (int off = 16; off > 0; off >>= 1) s += __shfl_xor(s, off, 32);
    float inv = 1.0f / s;
    __bf16* dst = c_mn + (size_t)tk * NEXP;
    #pragma unroll
    for (int j = 0; j < 4; ++j) dst[lane + j * 32] = (__bf16)(v[j] * inv);
}

// ---------------------------------------------------------------------------
// Kernel 6: xs[b] = d_nm[b] (128x1024) x xn[b] (1024x384) -> bf16 [B][N][D]
// grid (6, B); wave 32x32 tile: expert tile = w&3, d tile = blk.x*2 + (w>>2).
// ---------------------------------------------------------------------------
__global__ void k_gemm_slots(const __bf16* __restrict__ d_nm,
                             const __bf16* __restrict__ xn_dm,
                             __bf16* __restrict__ xs) {
    const int b    = blockIdx.y;
    const int w    = threadIdx.x >> 5;
    const int lane = threadIdx.x & 31;
    const int row  = lane & 15;
    const int half = lane >> 4;
    const int e0 = (w & 3) * 32;                     // expert rows (GEMM M)
    const int d0 = (blockIdx.x * 2 + (w >> 2)) * 32; // channel cols (GEMM N)
    const __bf16* a0p = d_nm  + ((size_t)(b * NEXP + e0 + row)) * MTOK + half * 8;
    const __bf16* a1p = a0p + (size_t)16 * MTOK;
    const __bf16* b0p = xn_dm + ((size_t)(b * DIM + d0 + row)) * MTOK + half * 8;
    const __bf16* b1p = b0p + (size_t)16 * MTOK;
    floatx8 acc00 = {}, acc01 = {}, acc10 = {}, acc11 = {};
    #pragma unroll 2
    for (int kk = 0; kk < MTOK; kk += 32) {
        bf16x16 a0 = load_k32(a0p + kk), a1 = load_k32(a1p + kk);
        bf16x16 b0 = load_k32(b0p + kk), b1 = load_k32(b1p + kk);
        acc00 = WMMA_BF16(a0, b0, acc00);
        acc01 = WMMA_BF16(a0, b1, acc01);
        acc10 = WMMA_BF16(a1, b0, acc10);
        acc11 = WMMA_BF16(a1, b1, acc11);
    }
    __bf16* d00 = xs + ((size_t)(b * NEXP + e0 + half * 8)) * DIM + d0 + row;
    store_tile_bf16(d00,                    DIM, acc00);
    store_tile_bf16(d00 + 16,               DIM, acc01);
    store_tile_bf16(d00 + (size_t)16 * DIM, DIM, acc10);
    store_tile_bf16(d00 + (size_t)16 * DIM + 16, DIM, acc11);
}

// ---------------------------------------------------------------------------
// Kernel 6b: convert expert weights to bf16 (vectorized f32x4 -> bf16x4).
// ---------------------------------------------------------------------------
__global__ void k_convert_w(const float* __restrict__ w, __bf16* __restrict__ wb,
                            int n4) {
    int i = blockIdx.x * 256 + threadIdx.x;
    if (i < n4) {
        float4 v = reinterpret_cast<const float4*>(w)[i];
        __bf16* d = wb + (size_t)i * 4;
        d[0] = (__bf16)v.x; d[1] = (__bf16)v.y;
        d[2] = (__bf16)v.z; d[3] = (__bf16)v.w;
    }
}

// ---------------------------------------------------------------------------
// Kernel 7: per-expert linear. ys_n (64x384) = xs[:,n,:] x W_n^T + b_n.
// grid (3, 128 experts); wave 32x32: batch tile = w&1, col tile = blk.x*4+(w>>1).
// Stored transposed as ys_dn[B][D][N] bf16 (K-contiguous for the final GEMM).
// ---------------------------------------------------------------------------
__global__ void k_gemm_expert(const __bf16* __restrict__ xs,
                              const __bf16* __restrict__ wb,
                              const float* __restrict__ b_exp,
                              __bf16* __restrict__ ys_dn) {
    const int n    = blockIdx.y;         // expert
    const int w    = threadIdx.x >> 5;
    const int lane = threadIdx.x & 31;
    const int row  = lane & 15;
    const int half = lane >> 4;
    const int b0 = (w & 1) * 32;                          // batch rows (GEMM M)
    const int o0 = (blockIdx.x * 4 + (w >> 1)) * 32;      // output cols (GEMM N)
    const __bf16* a0p = xs + ((size_t)((b0 + row) * NEXP + n)) * DIM + half * 8;
    const __bf16* a1p = a0p + (size_t)16 * NEXP * DIM;
    const __bf16* b0p = wb + (size_t)n * DIM * DIM + (size_t)(o0 + row) * DIM + half * 8;
    const __bf16* b1p = b0p + (size_t)16 * DIM;
    floatx8 acc00 = {}, acc01 = {}, acc10 = {}, acc11 = {};
    #pragma unroll 2
    for (int kk = 0; kk < DIM; kk += 32) {
        bf16x16 a0 = load_k32(a0p + kk), a1 = load_k32(a1p + kk);
        bf16x16 b0 = load_k32(b0p + kk), b1 = load_k32(b1p + kk);
        acc00 = WMMA_BF16(a0, b0, acc00);
        acc01 = WMMA_BF16(a0, b1, acc01);
        acc10 = WMMA_BF16(a1, b0, acc10);
        acc11 = WMMA_BF16(a1, b1, acc11);
    }
    const float bias0 = b_exp[(size_t)n * DIM + o0 + row];
    const float bias1 = b_exp[(size_t)n * DIM + o0 + 16 + row];
    #pragma unroll
    for (int r = 0; r < 8; ++r) {
        int bbL = b0 + r + half * 8;
        int bbH = bbL + 16;
        ys_dn[((size_t)(bbL * DIM + o0 + row)) * NEXP + n]      = (__bf16)(acc00[r] + bias0);
        ys_dn[((size_t)(bbL * DIM + o0 + 16 + row)) * NEXP + n] = (__bf16)(acc01[r] + bias1);
        ys_dn[((size_t)(bbH * DIM + o0 + row)) * NEXP + n]      = (__bf16)(acc10[r] + bias0);
        ys_dn[((size_t)(bbH * DIM + o0 + 16 + row)) * NEXP + n] = (__bf16)(acc11[r] + bias1);
    }
}

// ---------------------------------------------------------------------------
// Kernel 8: y[b] = c_mn[b] (1024x128) x ys[b] (128x384), written transposed
// into out[B][C][M] f32 through an LDS tile for a coalesced 100 MB store.
// grid (12, 4, B); wave 32x32 tile; block tile = 256 m x 32 d.
// ---------------------------------------------------------------------------
__global__ void k_gemm_combine(const __bf16* __restrict__ c_mn,
                               const __bf16* __restrict__ ys_dn,
                               float* __restrict__ out) {
    __shared__ float lds[32 * 258];
    const int b    = blockIdx.z;
    const int w    = threadIdx.x >> 5;
    const int lane = threadIdx.x & 31;
    const int row  = lane & 15;
    const int half = lane >> 4;
    const int m0 = blockIdx.y * 256 + w * 32;
    const int d0 = blockIdx.x * 32;
    const __bf16* a0p = c_mn  + ((size_t)(b * MTOK + m0 + row)) * NEXP + half * 8;
    const __bf16* a1p = a0p + (size_t)16 * NEXP;
    const __bf16* b0p = ys_dn + ((size_t)(b * DIM + d0 + row)) * NEXP + half * 8;
    const __bf16* b1p = b0p + (size_t)16 * NEXP;
    floatx8 acc00 = {}, acc01 = {}, acc10 = {}, acc11 = {};
    #pragma unroll
    for (int kk = 0; kk < NEXP; kk += 32) {
        bf16x16 a0 = load_k32(a0p + kk), a1 = load_k32(a1p + kk);
        bf16x16 b0 = load_k32(b0p + kk), b1 = load_k32(b1p + kk);
        acc00 = WMMA_BF16(a0, b0, acc00);
        acc01 = WMMA_BF16(a0, b1, acc01);
        acc10 = WMMA_BF16(a1, b0, acc10);
        acc11 = WMMA_BF16(a1, b1, acc11);
    }
    // lane holds (m_local = w*32 + mi*16 + r + half*8, d_local = ni*16 + row)
    #pragma unroll
    for (int r = 0; r < 8; ++r) {
        int mlL = w * 32 + r + half * 8;
        int mlH = mlL + 16;
        lds[(row)      * 258 + mlL] = acc00[r];
        lds[(16 + row) * 258 + mlL] = acc01[r];
        lds[(row)      * 258 + mlH] = acc10[r];
        lds[(16 + row) * 258 + mlH] = acc11[r];
    }
    __syncthreads();
    const int mbase = blockIdx.y * 256;
    for (int e = threadIdx.x; e < 32 * 256; e += 256) {
        int dl = e >> 8, ml = e & 255;
        out[((size_t)(b * DIM + d0 + dl)) * MTOK + mbase + ml] = lds[dl * 258 + ml];
    }
}

// ---------------------------------------------------------------------------
// Host-side launch. Inputs (f32): x, phi, scale, w_exp, b_exp. Output f32.
// ---------------------------------------------------------------------------
extern "C" void kernel_launch(void* const* d_in, const int* in_sizes, int n_in,
                              void* d_out, int out_size, void* d_ws, size_t ws_size,
                              hipStream_t stream) {
    const float* x     = (const float*)d_in[0];
    const float* phi   = (const float*)d_in[1];
    const float* scale = (const float*)d_in[2];
    const float* w_exp = (const float*)d_in[3];
    const float* b_exp = (const float*)d_in[4];
    float* out = (float*)d_out;

    char* ws = (char*)d_ws;
    size_t off = 0;
    auto take = [&](size_t bytes) -> char* {
        char* p = ws + off;
        off = (off + bytes + 255) & ~(size_t)255;
        return p;
    };
    const size_t sz_xn     = (size_t)BATCH * MTOK * DIM * 2;   // 50.3 MB each
    const size_t sz_logits = (size_t)BATCH * MTOK * NEXP * 4;  // 33.6 MB
    const size_t sz_dnm    = (size_t)BATCH * NEXP * MTOK * 2;  // 16.8 MB
    const size_t sz_cmn    = (size_t)BATCH * MTOK * NEXP * 2;  // 16.8 MB
    const size_t sz_xs     = (size_t)BATCH * NEXP * DIM * 2;   //  6.3 MB
    const size_t sz_ys     = (size_t)BATCH * DIM * NEXP * 2;   //  6.3 MB
    const size_t sz_part   = (size_t)BATCH * 16 * NEXP * 4;    //  0.5 MB
    const size_t sz_gvec   = (size_t)BATCH * NEXP * 4;         //  32 KB

    __bf16* xn_md  = (__bf16*)take(sz_xn);
    __bf16* xn_dm  = (__bf16*)take(sz_xn);
    __bf16* phin   = (__bf16*)take((size_t)NEXP * DIM * 2);
    float*  logits = (float*) take(sz_logits);
    __bf16* d_nm   = (__bf16*)take(sz_dnm);
    __bf16* c_mn   = (__bf16*)take(sz_cmn);
    __bf16* xs     = (__bf16*)take(sz_xs);
    __bf16* ys_dn  = (__bf16*)take(sz_ys);
    float*  cmax   = (float*) take(sz_part);
    float*  csum   = (float*) take(sz_part);
    float*  gmax   = (float*) take(sz_gvec);
    float*  ginv   = (float*) take(sz_gvec);
    // w_bf (37.7 MB) aliases xn_md (50.3 MB), which is dead after kernel 3.
    __bf16* w_bf   = xn_md;

    k_norm_tokens<<<dim3(MTOK / 32, BATCH), 256, 0, stream>>>(x, xn_md, xn_dm);
    k_norm_phi<<<1, NEXP, 0, stream>>>(phi, scale, phin);
    k_gemm_logits<<<dim3(NEXP / 32, MTOK / 256, BATCH), 256, 0, stream>>>(xn_md, phin, logits);
    {
        const int n4 = (NEXP * DIM * DIM) / 4;   // 4,718,592 float4s
        k_convert_w<<<(n4 + 255) / 256, 256, 0, stream>>>(w_exp, w_bf, n4);
    }
    k_smax_d_part<<<dim3(16, BATCH), NEXP, 0, stream>>>(logits, cmax, csum);
    k_smax_d_comb<<<BATCH, NEXP, 0, stream>>>(cmax, csum, gmax, ginv);
    k_smax_d_write<<<dim3(16, BATCH), 256, 0, stream>>>(logits, gmax, ginv, d_nm);
    k_softmax_combine<<<(BATCH * MTOK) / 8, 256, 0, stream>>>(logits, c_mn);
    k_gemm_slots<<<dim3(DIM / 64, BATCH), 256, 0, stream>>>(d_nm, xn_dm, xs);
    k_gemm_expert<<<dim3(DIM / 128, NEXP), 256, 0, stream>>>(xs, w_bf, b_exp, ys_dn);
    k_gemm_combine<<<dim3(DIM / 32, MTOK / 256, BATCH), 256, 0, stream>>>(c_mn, ys_dn, out);
}
